// Reservoir_11209864642805
// MI455X (gfx1250) — matrix operand order repaired and checked by reading, not compiled, not run on hardware
//
#include <hip/hip_runtime.h>

typedef __bf16 bf16_t;
typedef __attribute__((ext_vector_type(16))) __bf16 v16bf;
typedef __attribute__((ext_vector_type(8)))  __bf16 v8bf;
typedef __attribute__((ext_vector_type(8)))  float  v8f;

#define RES      1024
#define INP      16
#define OUTP     16
#define BATCHN   1024
#define SEQ      4096
#define MTILE    64                    // batch rows per group
#define NSPLIT   8                     // workgroups per group (N split)
#define NPER     (RES / NSPLIT)        // 128 reservoir rows per WG (W slice)
#define NGROUPS  (BATCHN / MTILE)      // 16 groups
#define NWG      (NGROUPS * NSPLIT)    // 128 workgroups
#define NTHREADS 512                   // 16 waves
#define KSTEPS   (RES / 32)

#if __has_builtin(__builtin_amdgcn_tanhf)
#define TANHF(v) __builtin_amdgcn_tanhf(v)
#elif __has_builtin(__builtin_amdgcn_tanh_f32)
#define TANHF(v) __builtin_amdgcn_tanh_f32(v)
#else
#define TANHF(v) tanhf(v)
#endif

// ---- one-shot: W f32 -> bf16 in workspace (async-copy source) ----
__global__ void convert_W_kernel(const float* __restrict__ W, bf16_t* __restrict__ Wbf) {
    int i = blockIdx.x * blockDim.x + threadIdx.x;   // RES*RES threads
    Wbf[i] = (bf16_t)W[i];
}

// Zero h(t=0) buffer and group barrier counters (runs every launch: replay-safe)
__global__ void init_kernel(bf16_t* __restrict__ hbuf, unsigned* __restrict__ counters) {
    size_t i = (size_t)blockIdx.x * blockDim.x + threadIdx.x;
    const size_t n0 = (size_t)NGROUPS * MTILE * RES;
    if (i < n0) hbuf[i] = (bf16_t)0.0f;
    if (i < NGROUPS) counters[i] = 0u;
}

__launch_bounds__(NTHREADS, 1)
__global__ void reservoir_kernel(const float* __restrict__ x,
                                 const float* __restrict__ Win,
                                 const bf16_t* __restrict__ Wbf,
                                 const float* __restrict__ Wout_w,
                                 const float* __restrict__ Wout_b,
                                 bf16_t* __restrict__ hbuf,       // [2][NGROUPS][MTILE][RES]
                                 unsigned* __restrict__ counters, // [NGROUPS]
                                 float* __restrict__ out) {
    // Persistent W slice: 128 rows x 1024 cols bf16 = 256 KB LDS (CDNA5 WGP LDS)
    __shared__ __align__(16) bf16_t wlds[NPER * RES];

    const int tid  = threadIdx.x;
    const int lane = tid & 31;
    const int wave = tid >> 5;          // 0..15
    const int l15  = lane & 15;
    const int sel  = lane >> 4;         // half-wave selector in WMMA layouts

    const int g = blockIdx.x / NSPLIT;  // batch group
    const int s = blockIdx.x % NSPLIT;  // N-slice within group

    const int nt = wave >> 1;           // 0..7 : N-tile within slice
    const int mh = wave & 1;            // 0/1  : owns m-tiles {2mh, 2mh+1}

    // ---- stage this WG's W slice into LDS via async copy (no VGPR round-trip) ----
    {
        const bf16_t* wsrc = Wbf + (size_t)s * NPER * RES;
        for (int i = tid * 8; i < NPER * RES; i += NTHREADS * 8) {   // 16 B per lane per op
            unsigned ldsAddr = (unsigned)(uintptr_t)&wlds[i];        // flat->LDS: addr[31:0]
            unsigned long long gAddr = (unsigned long long)(uintptr_t)(wsrc + i);
            asm volatile("global_load_async_to_lds_b128 %0, %1, off"
                         :: "v"(ldsAddr), "v"(gAddr) : "memory");
        }
        asm volatile("s_wait_asynccnt 0x0" ::: "memory");
    }

    // ---- Win B-fragment for this wave's n-tile (K padded 16->32) ----
    // B 32x16: lane -> column n=l15; elements 0-7 hold K=sel*8..sel*8+7, elements 8-15 K>=16 (zero)
    v16bf winfrag;
#pragma unroll
    for (int j = 0; j < 16; ++j) winfrag[j] = (bf16_t)0.0f;
    {
        const float* wp = Win + (size_t)(s * NPER + nt * 16 + l15) * INP;
#pragma unroll
        for (int j = 0; j < 8; ++j) winfrag[j] = (bf16_t)wp[sel * 8 + j];
    }

    // x row pointers for this wave's two m-tiles (A-layout half-wave offset folded in)
    const float* xrow[2];
    xrow[0] = x + (size_t)(g * MTILE + (2 * mh + 0) * 16 + l15) * SEQ * INP + sel * 8;
    xrow[1] = x + (size_t)(g * MTILE + (2 * mh + 1) * 16 + l15) * SEQ * INP + sel * 8;

    // per-lane LDS base for B fragments (element index)
    const int wldsBase = (nt * 16 + l15) * RES + sel * 16;

    bf16_t* hg = hbuf + (size_t)g * MTILE * RES;            // group base, buffer 0
    const size_t bufStride = (size_t)NGROUPS * MTILE * RES; // buffer 0 <-> 1
    unsigned* cnt = counters + g;

    __syncthreads();   // async W staging visible to all waves

    for (int t = 0; t < SEQ; ++t) {
        const int cur = t & 1;
        const bf16_t* hcur = hg + (size_t)cur * bufStride;
        bf16_t*       hnxt = hg + (size_t)(cur ^ 1) * bufStride;

        v8f acc[2];

        // ---- input injection: acc = x_t @ Win^T ----
#pragma unroll
        for (int j = 0; j < 2; ++j) {
            const float* xp = xrow[j] + (size_t)t * INP;
            v16bf xa;
#pragma unroll
            for (int q = 0; q < 8; ++q)  xa[q] = (bf16_t)xp[q];
#pragma unroll
            for (int q = 8; q < 16; ++q) xa[q] = (bf16_t)0.0f;
            v8f z = {};
            acc[j] = __builtin_amdgcn_wmma_f32_16x16x32_bf16(
                false, xa, false, winfrag, (short)0, z, false, false);
        }

        // prefetch next step's x while the K-loop runs (global_prefetch_b8)
        if (t + 1 < SEQ) {
            __builtin_prefetch(xrow[0] + (size_t)(t + 1) * INP, 0, 3);
            __builtin_prefetch(xrow[1] + (size_t)(t + 1) * INP, 0, 3);
        }

        // per-lane A row pointers into shared h (global, L2/WGP$)
        const bf16_t* ha0 = hcur + (size_t)((2 * mh + 0) * 16 + l15) * RES + sel * 16;
        const bf16_t* ha1 = hcur + (size_t)((2 * mh + 1) * 16 + l15) * RES + sel * 16;

        // ---- recurrent GEMM: acc += h @ W_slice^T, W from LDS ----
#pragma unroll 4
        for (int kk = 0; kk < KSTEPS; ++kk) {
            union { v16bf v; struct { v8bf lo, hi; } c; } a0, a1;
            a0.c.lo = *(const v8bf*)(ha0 + kk * 32);
            a0.c.hi = *(const v8bf*)(ha0 + kk * 32 + 8);
            a1.c.lo = *(const v8bf*)(ha1 + kk * 32);
            a1.c.hi = *(const v8bf*)(ha1 + kk * 32 + 8);
            v16bf bfrag = *(const v16bf*)&wlds[wldsBase + kk * 32];
            acc[0] = __builtin_amdgcn_wmma_f32_16x16x32_bf16(
                false, a0.v, false, bfrag, (short)0, acc[0], false, false);
            acc[1] = __builtin_amdgcn_wmma_f32_16x16x32_bf16(
                false, a1.v, false, bfrag, (short)0, acc[1], false, false);
        }

        // ---- h_next = tanh(acc) ----
        // C/D layout: VGPR i -> M = i + sel*8 (+16*mtile), N = l15 (+tile base)
        const int ncol = s * NPER + nt * 16 + l15;
#pragma unroll
        for (int j = 0; j < 2; ++j) {
            const int mrow = (2 * mh + j) * 16 + sel * 8;
#pragma unroll
            for (int i = 0; i < 8; ++i)
                hnxt[(size_t)(mrow + i) * RES + ncol] = (bf16_t)TANHF(acc[j][i]);
        }

        // ---- inter-WG barrier across the group's NSPLIT workgroups ----
        __builtin_amdgcn_fence(__ATOMIC_RELEASE, "agent");
        __syncthreads();
        if (tid == 0) {
            __hip_atomic_fetch_add(cnt, 1u, __ATOMIC_RELAXED, __HIP_MEMORY_SCOPE_AGENT);
            const unsigned target = (unsigned)(t + 1) * NSPLIT;
            while (__hip_atomic_load(cnt, __ATOMIC_RELAXED, __HIP_MEMORY_SCOPE_AGENT) < target)
                __builtin_amdgcn_s_sleep(2);
        }
        __syncthreads();
        __builtin_amdgcn_fence(__ATOMIC_ACQUIRE, "agent");  // global_inv: fresh h next step
    }

    // ---- readout: y = h_final @ Wout^T + b (final h is in buffer 0; SEQ even) ----
    if (s == 0) {
        for (int idx = tid; idx < MTILE * OUTP; idx += NTHREADS) {
            const int m = idx >> 4, o = idx & 15;
            float sum = Wout_b[o];
            const float*  wr = Wout_w + (size_t)o * RES;
            const bf16_t* hr = hg + (size_t)m * RES;
            for (int k = 0; k < RES; ++k) sum += (float)hr[k] * wr[k];
            out[(size_t)(g * MTILE + m) * OUTP + o] = sum;
        }
    }
}

extern "C" void kernel_launch(void* const* d_in, const int* in_sizes, int n_in,
                              void* d_out, int out_size, void* d_ws, size_t ws_size,
                              hipStream_t stream) {
    (void)in_sizes; (void)n_in; (void)out_size; (void)ws_size;

    const float* x      = (const float*)d_in[0];
    const float* Win    = (const float*)d_in[1];
    const float* W      = (const float*)d_in[2];
    const float* Wout_w = (const float*)d_in[3];
    const float* Wout_b = (const float*)d_in[4];
    float* out = (float*)d_out;

    // ws layout: [0,4MB) h double buffer | [4MB,6MB) Wbf | [6MB,+64) counters
    const size_t hElems = (size_t)2 * NGROUPS * MTILE * RES;
    bf16_t*   hbuf     = (bf16_t*)d_ws;
    bf16_t*   Wbf      = (bf16_t*)((char*)d_ws + hElems * sizeof(bf16_t));
    unsigned* counters = (unsigned*)((char*)d_ws + (hElems + (size_t)RES * RES) * sizeof(bf16_t));

    convert_W_kernel<<<(RES * RES) / 256, 256, 0, stream>>>(W, Wbf);

    const size_t initN = (size_t)NGROUPS * MTILE * RES;
    init_kernel<<<(int)((initN + 255) / 256), 256, 0, stream>>>(hbuf, counters);

    reservoir_kernel<<<NWG, NTHREADS, 0, stream>>>(x, Win, Wbf, Wout_w, Wout_b,
                                                   hbuf, counters, out);
}
